// ParallelBlockMultiHeadAttention_24842090840768
// MI455X (gfx1250) — matrix operand order repaired and checked by reading, not compiled, not run on hardware
//
#include <hip/hip_runtime.h>
#include <math.h>

#define DIMC   768
#define NHEAD  12
#define HDIM   64
#define SEQ    2048
#define BATCH  4
#define QROWS  128          // query rows per workgroup (8 waves x 16)
#define KBLK   32           // key rows per iteration
#define NTHREADS 256
#define QSTR   72           // LDS row stride (halfs), 144B = 9*16
#define KSTR   72
#define VSTR   40           // 80B = 5*16
#define SM_C   0.18033688011112042f   // HDIM^-0.5 * log2(e)  (softmax in exp2 domain)
#define LN_EPS 1e-6f

typedef __attribute__((ext_vector_type(16))) _Float16 v16h;
typedef __attribute__((ext_vector_type(8)))  _Float16 h8;
typedef __attribute__((ext_vector_type(8)))  float    v8f;

#if defined(__gfx1250__) && __has_builtin(__builtin_amdgcn_global_load_async_to_lds_b128)
#define HAVE_ASYNC 1
typedef int v4i_ __attribute__((vector_size(16)));
typedef __attribute__((address_space(3))) v4i_ lds_v4i;
typedef __attribute__((address_space(1))) v4i_ gbl_v4i;
#else
#define HAVE_ASYNC 0
#endif

// ---- 16B global -> LDS transfer: async DMA (ASYNCcnt) when available ----
__device__ __forceinline__ void ld16_to_lds(const float* g, float* l) {
#if HAVE_ASYNC
    __builtin_amdgcn_global_load_async_to_lds_b128((gbl_v4i*)g, (lds_v4i*)l, 0, 0);
#else
    *(float4*)l = *(const float4*)g;
#endif
}

__device__ __forceinline__ void wait_async0() {
#if HAVE_ASYNC
#if __has_builtin(__builtin_amdgcn_s_wait_asynccnt)
    __builtin_amdgcn_s_wait_asynccnt(0);
#else
    asm volatile("s_wait_asynccnt 0x0" ::: "memory");
#endif
#endif
}

// ---- lane-half swap (lane ^ 16) as pure VALU via v_permlanex16_b32 ----
__device__ __forceinline__ float xor16f(float x) {
#if defined(__gfx1250__) && __has_builtin(__builtin_amdgcn_permlanex16)
    int r = __builtin_amdgcn_permlanex16(__float_as_int(x), __float_as_int(x),
                                         0x76543210, 0xfedcba98, false, false);
    return __int_as_float(r);
#else
    return __shfl_xor(x, 16, 32);
#endif
}

__device__ __forceinline__ float fexp2(float x) {
#if defined(__gfx1250__) && __has_builtin(__builtin_amdgcn_exp2f)
    return __builtin_amdgcn_exp2f(x);       // v_exp_f32
#else
    return exp2f(x);
#endif
}

__device__ __forceinline__ v16h make_v16(h8 lo, h8 hi) {
    return __builtin_shufflevector(lo, hi, 0,1,2,3,4,5,6,7,8,9,10,11,12,13,14,15);
}

// A-operand (16-bit A 16x32 layout): lane<16 holds K {cb..cb+15},
// lane>=16 holds K {cb+16..cb+31} (contiguous); row = M for that lane half.
__device__ __forceinline__ v16h lds_A(const _Float16* t, int row, int stride,
                                      int cb, int h8off) {
    const h8* p0 = (const h8*)(t + row * stride + cb + 2 * h8off);
    const h8* p1 = (const h8*)(t + row * stride + cb + 8 + 2 * h8off);
    return make_v16(*p0, *p1);
}

// B-operand (16-bit B 32x16 layout): lane<16 = col lane, K=cb..cb+15 (contiguous);
// lane>=16 = col lane-16, K=cb+16..cb+31 (caller folds +16 into cb).
__device__ __forceinline__ v16h lds_B(const _Float16* t, int row, int stride, int cb) {
    const h8* p0 = (const h8*)(t + row * stride + cb);
    return make_v16(p0[0], p0[1]);
}

__global__ __launch_bounds__(NTHREADS)
void fused_ln_flash_attn(const float* __restrict__ qkv,
                         const float* __restrict__ q_scale, const float* __restrict__ q_bias,
                         const float* __restrict__ k_scale, const float* __restrict__ k_bias,
                         float* __restrict__ out)
{
    __shared__ __align__(16) _Float16 sQ[QROWS * QSTR];   // 18.0 KB
    __shared__ __align__(16) _Float16 sK[KBLK  * KSTR];   //  4.5 KB
    __shared__ __align__(16) _Float16 sV[HDIM  * VSTR];   //  5.0 KB (transposed: [d][k])
    __shared__ __align__(16) float rawK[KBLK * HDIM];     //  8.0 KB async staging
    __shared__ __align__(16) float rawV[KBLK * HDIM];     //  8.0 KB async staging
    __shared__ float redS[QROWS], redSS[QROWS];
    __shared__ float gq[HDIM], bq[HDIM], gk[HDIM], bk[HDIM];

    const int tid  = threadIdx.x;
    const int lane = tid & 31;
    const int wave = tid >> 5;

    const int nqc = SEQ / QROWS;            // 16
    const int qc  = blockIdx.x % nqc;
    const int bh  = blockIdx.x / nqc;
    const int h   = bh % NHEAD;
    const int b   = bh / NHEAD;

    const int hi8  = (lane & 16) ? 8  : 0;  // A-layout lane-half K offset
    const int hi16 = (lane & 16) ? 16 : 0;  // B-layout lane-half K offset
    const int l15  = lane & 15;

    const size_t rowBase = (size_t)b * SEQ * (3 * DIMC);
    const int hcol = h * HDIM;
    const int col  = (tid & 15) * 4;        // 16 lanes cover one 64-float row

    // ---- kick off async staging of K/V block 0 (overlaps all of Q setup) ----
    {
#pragma unroll
        for (int i = 0; i < 2; ++i) {
            int idx = i * NTHREADS + tid;
            int row = idx >> 4;             // 0..31
            const float* gkp = qkv + rowBase + (size_t)row * (3 * DIMC) + DIMC + hcol + col;
            ld16_to_lds(gkp,        rawK + idx * 4);
            ld16_to_lds(gkp + DIMC, rawV + idx * 4);
        }
    }

    if (tid < HDIM) { gq[tid] = q_scale[tid]; bq[tid] = q_bias[tid];
                      gk[tid] = k_scale[tid]; bk[tid] = k_bias[tid]; }
    if (tid < QROWS) { redS[tid] = 0.f; redSS[tid] = 0.f; }
    __syncthreads();

    // ---- stage Q: coalesced float4 loads, LN stats via ds_add_f32, f16 store ----
    float4 qraw[8];
#pragma unroll
    for (int i = 0; i < 8; ++i) {
        int idx = i * NTHREADS + tid;
        int row = idx >> 4;                 // 0..127
        float4 v = *(const float4*)(qkv + rowBase
                     + (size_t)(qc * QROWS + row) * (3 * DIMC) + hcol + col);
        qraw[i] = v;
        atomicAdd(&redS[row],  v.x + v.y + v.z + v.w);
        atomicAdd(&redSS[row], v.x*v.x + v.y*v.y + v.z*v.z + v.w*v.w);
    }
    __syncthreads();
#pragma unroll
    for (int i = 0; i < 8; ++i) {
        int idx = i * NTHREADS + tid;
        int row = idx >> 4;
        float mean = redS[row]  * (1.f / HDIM);
        float var  = redSS[row] * (1.f / HDIM) - mean * mean;
        float rstd = rsqrtf(var + LN_EPS);
        const float* pv = (const float*)&qraw[i];
#pragma unroll
        for (int j = 0; j < 4; ++j) {
            float y = (pv[j] - mean) * rstd * gq[col + j] + bq[col + j];
            sQ[row * QSTR + col + j] = (_Float16)y;
        }
    }
    __syncthreads();                        // Q tile ready; all red reads done
    if (tid < KBLK) { redS[tid] = 0.f; redSS[tid] = 0.f; }
    wait_async0();
    __syncthreads();                        // raw block 0 in LDS; stats cleared

    // ---- per-lane flash-attention state (softmax kept in log2 domain) ----
    float m_run = -3.0e38f, l_run = 0.f;
    v8f o0 = {0,0,0,0,0,0,0,0}, o1 = o0, o2 = o0, o3 = o0;
    const int qrow = wave * 16 + l15;       // row in sQ for B operand

    for (int kb0 = 0; kb0 < SEQ; kb0 += KBLK) {
        // step1: consume raw panel: K stats + V transpose/convert
        float4 kraw[2];
#pragma unroll
        for (int i = 0; i < 2; ++i) {
            int idx = i * NTHREADS + tid;
            int row = idx >> 4;             // 0..31
            float4 kv = *(const float4*)(rawK + idx * 4);
            kraw[i] = kv;
            atomicAdd(&redS[row],  kv.x + kv.y + kv.z + kv.w);
            atomicAdd(&redSS[row], kv.x*kv.x + kv.y*kv.y + kv.z*kv.z + kv.w*kv.w);
            float4 vv = *(const float4*)(rawV + idx * 4);
            sV[(col + 0) * VSTR + row] = (_Float16)vv.x;
            sV[(col + 1) * VSTR + row] = (_Float16)vv.y;
            sV[(col + 2) * VSTR + row] = (_Float16)vv.z;
            sV[(col + 3) * VSTR + row] = (_Float16)vv.w;
        }
        __syncthreads();                    // stats done; raw panel fully read

        // step3: prefetch next block into raw staging (overlaps LN + WMMA)
        if (kb0 + KBLK < SEQ) {
#pragma unroll
            for (int i = 0; i < 2; ++i) {
                int idx = i * NTHREADS + tid;
                int row = idx >> 4;
                const float* gkp = qkv + rowBase
                    + (size_t)(kb0 + KBLK + row) * (3 * DIMC) + DIMC + hcol + col;
                ld16_to_lds(gkp,        rawK + idx * 4);
                ld16_to_lds(gkp + DIMC, rawV + idx * 4);
            }
        }

        // step4: layernorm K, write f16 tile
#pragma unroll
        for (int i = 0; i < 2; ++i) {
            int idx = i * NTHREADS + tid;
            int row = idx >> 4;
            float mean = redS[row]  * (1.f / HDIM);
            float var  = redSS[row] * (1.f / HDIM) - mean * mean;
            float rstd = rsqrtf(var + LN_EPS);
            const float* pv = (const float*)&kraw[i];
#pragma unroll
            for (int j = 0; j < 4; ++j) {
                float y = (pv[j] - mean) * rstd * gk[col + j] + bk[col + j];
                sK[row * KSTR + col + j] = (_Float16)y;
            }
        }
        __syncthreads();                    // f16 K/V tiles ready
        if (tid < KBLK) { redS[tid] = 0.f; redSS[tid] = 0.f; }  // for next iter

        // ---- GEMM1: S^T (32k x 16q) = K_blk x Q^T, accumulate over d=64 ----
        v8f c1 = {0,0,0,0,0,0,0,0}, c2 = c1;
#pragma unroll
        for (int dk = 0; dk < HDIM; dk += 32) {
            v16h bQ = lds_B(sQ, qrow, QSTR, dk + hi16);
            v16h a1 = lds_A(sK, l15,      KSTR, dk, hi8);
            c1 = __builtin_amdgcn_wmma_f32_16x16x32_f16(false, a1, false, bQ,
                                                        (short)0, c1, false, false);
            v16h a2 = lds_A(sK, 16 + l15, KSTR, dk, hi8);
            c2 = __builtin_amdgcn_wmma_f32_16x16x32_f16(false, a2, false, bQ,
                                                        (short)0, c2, false, false);
        }

        // ---- online softmax (k axis = 8 VGPRs + lane^16), exp2 domain ----
        float p1[8], p2[8];
        float mb = -3.0e38f;
#pragma unroll
        for (int v = 0; v < 8; ++v) {
            p1[v] = c1[v] * SM_C; p2[v] = c2[v] * SM_C;
            mb = fmaxf(mb, fmaxf(p1[v], p2[v]));
        }
        mb = fmaxf(mb, xor16f(mb));
        float mnew  = fmaxf(m_run, mb);
        float alpha = fexp2(m_run - mnew);
        float lb = 0.f;
#pragma unroll
        for (int v = 0; v < 8; ++v) {
            p1[v] = fexp2(p1[v] - mnew);
            p2[v] = fexp2(p2[v] - mnew);
            lb += p1[v] + p2[v];
        }
        lb += xor16f(lb);
        l_run = l_run * alpha + lb;
        m_run = mnew;
#pragma unroll
        for (int v = 0; v < 8; ++v) {
            o0[v] *= alpha; o1[v] *= alpha; o2[v] *= alpha; o3[v] *= alpha;
        }

        // ---- build B = P^T (32k x 16q) in registers (v_permlanex16) ----
        float t1[8], t2[8];
#pragma unroll
        for (int v = 0; v < 8; ++v) { t1[v] = xor16f(p1[v]); t2[v] = xor16f(p2[v]); }
        const bool loh = (lane < 16);
        v16h bP = {};
#pragma unroll
        for (int i = 0; i < 4; ++i) {       // K=2i,2i+1 (lanes<16) / 16+2i.. (lanes>=16)
            bP[2*i]   = (_Float16)(loh ? p1[2*i]   : t2[2*i]);
            bP[2*i+1] = (_Float16)(loh ? p1[2*i+1] : t2[2*i+1]);
        }
#pragma unroll
        for (int i = 4; i < 8; ++i) {
            int j = 2*i - 8;
            bP[2*i]   = (_Float16)(loh ? t1[j]   : p2[j]);
            bP[2*i+1] = (_Float16)(loh ? t1[j+1] : p2[j+1]);
        }

        // ---- GEMM2: O^T (64d x 16q) += V^T (64x32) x P^T (32x16) ----
        v16h a;
        a  = lds_A(sV,  0 + l15, VSTR, 0, hi8);
        o0 = __builtin_amdgcn_wmma_f32_16x16x32_f16(false, a, false, bP, (short)0, o0, false, false);
        a  = lds_A(sV, 16 + l15, VSTR, 0, hi8);
        o1 = __builtin_amdgcn_wmma_f32_16x16x32_f16(false, a, false, bP, (short)0, o1, false, false);
        a  = lds_A(sV, 32 + l15, VSTR, 0, hi8);
        o2 = __builtin_amdgcn_wmma_f32_16x16x32_f16(false, a, false, bP, (short)0, o2, false, false);
        a  = lds_A(sV, 48 + l15, VSTR, 0, hi8);
        o3 = __builtin_amdgcn_wmma_f32_16x16x32_f16(false, a, false, bP, (short)0, o3, false, false);

        // step8: next raw panel landed; compute done -> tiles reusable
        wait_async0();
        __syncthreads();
    }

    // ---- epilogue: normalize and store out[b][h][q][d] ----
    float inv = 1.f / l_run;
    int qg = qc * QROWS + wave * 16 + l15;
    size_t obase = (((size_t)b * NHEAD + h) * SEQ + qg) * HDIM;
    v8f oo[4] = {o0, o1, o2, o3};
#pragma unroll
    for (int t = 0; t < 4; ++t) {
        int d0 = t * 16 + hi8;              // C layout: d = vgpr + lane-half*8 + 16*t
        float4 s0, s1;
        s0.x = oo[t][0]*inv; s0.y = oo[t][1]*inv; s0.z = oo[t][2]*inv; s0.w = oo[t][3]*inv;
        s1.x = oo[t][4]*inv; s1.y = oo[t][5]*inv; s1.z = oo[t][6]*inv; s1.w = oo[t][7]*inv;
        *(float4*)(out + obase + d0)     = s0;
        *(float4*)(out + obase + d0 + 4) = s1;
    }
}

extern "C" void kernel_launch(void* const* d_in, const int* in_sizes, int n_in,
                              void* d_out, int out_size, void* d_ws, size_t ws_size,
                              hipStream_t stream) {
    const float* qkv = (const float*)d_in[0];
    const float* qs  = (const float*)d_in[1];
    const float* qb  = (const float*)d_in[2];
    const float* ks  = (const float*)d_in[3];
    const float* kb  = (const float*)d_in[4];
    float* out = (float*)d_out;
    dim3 grid(BATCH * NHEAD * (SEQ / QROWS));   // 768 workgroups
    fused_ln_flash_attn<<<grid, NTHREADS, 0, stream>>>(qkv, qs, qb, ks, kb, out);
}
